// MambaBlock_4475355922525
// MI455X (gfx1250) — matrix operand order repaired and checked by reading, not compiled, not run on hardware
//
#include <hip/hip_runtime.h>
#include <hip/hip_bf16.h>

// ---------------- types ----------------
typedef __bf16 bf16;
typedef __attribute__((ext_vector_type(16))) __bf16 v16bf;
typedef __attribute__((ext_vector_type(8)))  __bf16 v8bf;
typedef __attribute__((ext_vector_type(8)))  float  v8f;

// problem constants (match reference)
#define BSZ 2
#define TLEN 2048
#define HDIM 1024
#define IDIM 2048
#define NSTATE 16
#define RRANK 64
#define KCONV 4
#define PDIM (RRANK + 2 * NSTATE)   // 96
#define MROWS (BSZ * TLEN)          // 4096

// GEMM tiling: block = 256 threads (8 waves), tile 64(M) x 128(N), K step 32
#define BM 64
#define BN 128
#define BKK 32
#define LDSS 40   // LDS row stride in bf16 elements (80B, multiple of 16B)

// epilogue modes
#define EPI_F32       0
#define EPI_BF16      1
#define EPI_SPLITSILU 2
#define EPI_SOFTPLUS  3

// ---------------- fp32 -> bf16 convert ----------------
__global__ __launch_bounds__(256) void cvt_bf16_kernel(const float* __restrict__ in,
                                                       bf16* __restrict__ out, long n) {
    long i = (long)blockIdx.x * 256 + threadIdx.x;
    if (i < n) out[i] = (bf16)in[i];
}

// ---------------- RMSNorm + bf16 cast ----------------
__global__ __launch_bounds__(256) void rmsnorm_kernel(const float* __restrict__ x,
                                                      const float* __restrict__ scale,
                                                      bf16* __restrict__ xn) {
    const long row = blockIdx.x;                 // B*T rows
    const float* xr = x + row * (long)HDIM;
    float ss = 0.f;
    for (int h = threadIdx.x; h < HDIM; h += 256) { float v = xr[h]; ss += v * v; }
    #pragma unroll
    for (int off = 16; off > 0; off >>= 1) ss += __shfl_xor(ss, off, 32);
    __shared__ float wsum[8];
    if ((threadIdx.x & 31) == 0) wsum[threadIdx.x >> 5] = ss;
    __syncthreads();
    float tot = 0.f;
    #pragma unroll
    for (int w = 0; w < 8; ++w) tot += wsum[w];
    const float rs = rsqrtf(tot / (float)HDIM + 1e-6f);
    bf16* xo = xn + row * (long)HDIM;
    for (int h = threadIdx.x; h < HDIM; h += 256) xo[h] = (bf16)(xr[h] * rs * scale[h]);
}

// ---------------- bf16 WMMA GEMM: C = A(MxK) @ B(KxN) (+bias) + epilogue ----------------
// NGUARD=false requires N % BN == 0.
template <bool NGUARD>
__global__ __launch_bounds__(256) void gemm_bf16_kernel(
    const bf16* __restrict__ A, const bf16* __restrict__ Bmat,
    long M, long N, long K,
    const float* __restrict__ bias,
    float* __restrict__ outF, bf16* __restrict__ outB,
    int mode, long ldo, long split) {

    __shared__ __align__(16) bf16 As[BM * LDSS];   // 5120B
    __shared__ __align__(16) bf16 Bs[BN * LDSS];   // 10240B, transposed: Bs[n][k]

    const int tid  = threadIdx.x;
    const long m0  = (long)blockIdx.y * BM;
    const long n0  = (long)blockIdx.x * BN;

    const int wave = tid >> 5;        // 0..7
    const int lane = tid & 31;
    const int wm   = wave >> 1;       // 0..3  (16-row M tiles)
    const int wn   = wave & 1;        // 0..1  (64-col N tiles)
    const int half = lane >> 4;       // hi/lo half of wave
    const int lrow = lane & 15;

    v8f acc[4] = {{}, {}, {}, {}};

    // staging maps
    const int a_chunk = tid & 3;      // 4 chunks of 8 bf16 along K
    const int a_row   = tid >> 2;     // 0..63
    const int b_nc    = tid & 15;     // 16 chunks of 8 along N
    const int b_kb    = tid >> 4;     // 0..15 (covers k with kk*16)

    for (long k0 = 0; k0 < K; k0 += BKK) {
        __syncthreads();

        // ---- stage A 64x32 via async copy to LDS (one b128 per lane) ----
        {
            const bf16* gsrc = A + (m0 + a_row) * K + k0 + a_chunk * 8;
            unsigned lds_addr = (unsigned)(uintptr_t)(&As[a_row * LDSS + a_chunk * 8]);
            asm volatile("global_load_async_to_lds_b128 %0, %1, off"
                         :: "v"(lds_addr), "v"(gsrc) : "memory");
        }

        // ---- stage B 32x128 transposed into Bs[n][k] ----
        #pragma unroll
        for (int kk = 0; kk < 2; ++kk) {
            const int k = b_kb + kk * 16;
            if (NGUARD) {
                #pragma unroll
                for (int j = 0; j < 8; ++j) {
                    long n = n0 + b_nc * 8 + j;
                    if (n >= N) n = N - 1;   // clamp: padded cols never stored
                    Bs[(b_nc * 8 + j) * LDSS + k] = Bmat[(k0 + k) * N + n];
                }
            } else {
                const bf16* bsrc = Bmat + (k0 + k) * N + n0 + b_nc * 8;
                v8bf bv = *(const v8bf*)bsrc;
                #pragma unroll
                for (int j = 0; j < 8; ++j)
                    Bs[(b_nc * 8 + j) * LDSS + k] = bv[j];
            }
        }

        asm volatile("s_wait_asynccnt 0x0" ::: "memory");
        __syncthreads();

        // ---- A fragment per ISA 16-bit 16x32 layout ----
        v16bf afrag;
        {
            const int row = wm * 16 + lrow;
            const int kb  = half * 8;
            v8bf lo = *(const v8bf*)(&As[row * LDSS + kb]);
            v8bf hi = *(const v8bf*)(&As[row * LDSS + kb + 16]);
            #pragma unroll
            for (int j = 0; j < 8; ++j) { afrag[j] = lo[j]; afrag[8 + j] = hi[j]; }
        }
        // ---- 4 B fragments + WMMAs ----
        #pragma unroll
        for (int s = 0; s < 4; ++s) {
            const int n  = wn * 64 + s * 16 + lrow;
            const int kb = half * 8;
            v8bf lo = *(const v8bf*)(&Bs[n * LDSS + kb]);
            v8bf hi = *(const v8bf*)(&Bs[n * LDSS + kb + 16]);
            v16bf bfrag;
            #pragma unroll
            for (int j = 0; j < 8; ++j) { bfrag[j] = lo[j]; bfrag[8 + j] = hi[j]; }
            acc[s] = __builtin_amdgcn_wmma_f32_16x16x32_bf16(false, afrag, false, bfrag,
                                                             (short)0, acc[s], false, false);
        }
    }

    // epilogue; C/D layout: VGPR r -> M = half*8 + r, N = lrow
    #pragma unroll
    for (int s = 0; s < 4; ++s) {
        const long n = n0 + wn * 64 + s * 16 + lrow;
        if (NGUARD && n >= N) continue;
        #pragma unroll
        for (int r = 0; r < 8; ++r) {
            const long m = m0 + wm * 16 + half * 8 + r;
            float v = acc[s][r];
            if (bias) v += bias[n];
            if (mode == EPI_F32) {
                outF[m * ldo + n] = v;
            } else if (mode == EPI_BF16) {
                outB[m * ldo + n] = (bf16)v;
            } else if (mode == EPI_SPLITSILU) {
                if (n < split) outB[m * split + n] = (bf16)v;                          // hidden
                else           outF[m * split + (n - split)] = v / (1.f + __expf(-v)); // silu(gate)
            } else { // EPI_SOFTPLUS
                outF[m * ldo + n] = (v > 20.f) ? v : log1pf(__expf(v));
            }
        }
    }
}

// ---------------- depthwise causal conv K=4 + SiLU ----------------
__global__ __launch_bounds__(256) void conv_silu_kernel(const bf16* __restrict__ hidden,
                                                        const float* __restrict__ conv_w,
                                                        const float* __restrict__ conv_b,
                                                        float* __restrict__ convF,
                                                        bf16* __restrict__ convB) {
    const long idx = (long)blockIdx.x * 256 + threadIdx.x;   // over B*T*I
    const int  i  = (int)(idx % IDIM);
    const long bt = idx / IDIM;
    const int  t  = (int)(bt % TLEN);
    const long b  = bt / TLEN;
    float acc = conv_b[i];
    #pragma unroll
    for (int k = 0; k < KCONV; ++k) {
        const int tt = t - (KCONV - 1) + k;
        if (tt >= 0)
            acc += conv_w[k * IDIM + i] * (float)hidden[(b * TLEN + tt) * (long)IDIM + i];
    }
    const float s = acc / (1.f + __expf(-acc));
    convF[idx] = s;
    convB[idx] = (bf16)s;
}

// ---------------- extract ts (cols 0..R-1 of xpj) as bf16 ----------------
__global__ __launch_bounds__(256) void ts_extract_kernel(const float* __restrict__ xpj,
                                                         bf16* __restrict__ ts) {
    const long idx = (long)blockIdx.x * 256 + threadIdx.x;   // over M*R
    const long m = idx / RRANK;
    const int  r = (int)(idx % RRANK);
    ts[m * RRANK + r] = (bf16)xpj[m * PDIM + r];
}

// ---------------- sequential SSM scan over T ----------------
// one lane per (b,i,n); 16-lane shuffle reduction over n; lane n==0 writes y*gate
__global__ __launch_bounds__(256) void scan_kernel(const float* __restrict__ convF,
                                                   const float* __restrict__ delta,
                                                   const float* __restrict__ gate,
                                                   const float* __restrict__ xpj,
                                                   const float* __restrict__ A_log,
                                                   const float* __restrict__ Dvec,
                                                   bf16* __restrict__ ssmB) {
    const int tid = blockIdx.x * 256 + threadIdx.x;   // B*I*N lanes
    const int n = tid & (NSTATE - 1);
    const int c = tid >> 4;                           // 0..B*I-1
    const int i = c % IDIM;
    const int b = c / IDIM;

    const float Aval = -__expf(A_log[i * NSTATE + n]);
    const float Dv   = Dvec[i];
    float state = 0.f;

    const long baseCh = ((long)b * TLEN) * IDIM + i;
    const long baseP  = ((long)b * TLEN) * PDIM;

    for (int t = 0; t < TLEN; ++t) {
        const long o = baseCh + (long)t * IDIM;
        const float d = delta[o];
        const float h = convF[o];
        const float g = gate[o];
        const float* p = xpj + baseP + (long)t * PDIM;
        const float Bv = p[RRANK + n];
        const float Cv = p[RRANK + NSTATE + n];

        state = __expf(Aval * d) * state + d * Bv * h;

        float part = state * Cv;
        #pragma unroll
        for (int off = 8; off; off >>= 1) part += __shfl_xor(part, off, 32);

        if (n == 0) ssmB[o] = (bf16)((part + Dv * h) * g);
    }
}

// ---------------- host orchestration ----------------
extern "C" void kernel_launch(void* const* d_in, const int* in_sizes, int n_in,
                              void* d_out, int out_size, void* d_ws, size_t ws_size,
                              hipStream_t stream) {
    (void)in_sizes; (void)n_in; (void)out_size; (void)ws_size;

    const float* x          = (const float*)d_in[0];
    const float* norm_scale = (const float*)d_in[1];
    const float* in_w       = (const float*)d_in[2];
    const float* in_b       = (const float*)d_in[3];
    const float* conv_w     = (const float*)d_in[4];
    const float* conv_b     = (const float*)d_in[5];
    const float* xproj_w    = (const float*)d_in[6];
    const float* dt_w       = (const float*)d_in[7];
    const float* dt_b       = (const float*)d_in[8];
    const float* A_log      = (const float*)d_in[9];
    const float* Dvec       = (const float*)d_in[10];
    const float* out_w      = (const float*)d_in[11];
    const float* out_b      = (const float*)d_in[12];
    float*       out        = (float*)d_out;

    const long M = MROWS;

    // workspace layout (256B aligned cursor)
    char* ws = (char*)d_ws;
    size_t off = 0;
    auto alloc = [&](size_t bytes) { char* p = ws + off; off = (off + bytes + 255) & ~(size_t)255; return p; };

    bf16*  xn_bf     = (bf16*) alloc((size_t)M * HDIM * 2);
    bf16*  in_wb     = (bf16*) alloc((size_t)HDIM * 2 * IDIM * 2);
    bf16*  hidden_bf = (bf16*) alloc((size_t)M * IDIM * 2);
    float* gate_f    = (float*)alloc((size_t)M * IDIM * 4);
    float* conv_f    = (float*)alloc((size_t)M * IDIM * 4);
    bf16*  conv_bf   = (bf16*) alloc((size_t)M * IDIM * 2);
    bf16*  xproj_wb  = (bf16*) alloc((size_t)IDIM * PDIM * 2);
    float* xpj       = (float*)alloc((size_t)M * PDIM * 4);
    bf16*  ts_bf     = (bf16*) alloc((size_t)M * RRANK * 2);
    bf16*  dt_wb     = (bf16*) alloc((size_t)RRANK * IDIM * 2);
    float* delta_f   = (float*)alloc((size_t)M * IDIM * 4);
    bf16*  ssm_bf    = (bf16*) alloc((size_t)M * IDIM * 2);
    bf16*  out_wb    = (bf16*) alloc((size_t)IDIM * HDIM * 2);

    auto cvt = [&](const float* src, bf16* dst, long n) {
        cvt_bf16_kernel<<<dim3((unsigned)((n + 255) / 256)), dim3(256), 0, stream>>>(src, dst, n);
    };
    // weight conversions (fp32 -> bf16)
    cvt(in_w,    in_wb,    (long)HDIM * 2 * IDIM);
    cvt(xproj_w, xproj_wb, (long)IDIM * PDIM);
    cvt(dt_w,    dt_wb,    (long)RRANK * IDIM);
    cvt(out_w,   out_wb,   (long)IDIM * HDIM);

    // 1) RMSNorm
    rmsnorm_kernel<<<dim3((unsigned)M), dim3(256), 0, stream>>>(x, norm_scale, xn_bf);

    // 2) in-proj GEMM: [M,H]@[H,2I] -> hidden(bf16) | silu(gate)(f32)
    gemm_bf16_kernel<false><<<dim3((unsigned)(2 * IDIM / BN), (unsigned)(M / BM)),
                             dim3(256), 0, stream>>>(
        xn_bf, in_wb, M, 2 * IDIM, HDIM, in_b,
        gate_f, hidden_bf, EPI_SPLITSILU, 2 * IDIM, IDIM);

    // 3) depthwise conv + SiLU
    conv_silu_kernel<<<dim3((unsigned)(M * IDIM / 256)), dim3(256), 0, stream>>>(
        hidden_bf, conv_w, conv_b, conv_f, conv_bf);

    // 4) x-proj GEMM: [M,I]@[I,96] -> xpj (f32), no bias  (N-guarded)
    gemm_bf16_kernel<true><<<dim3((unsigned)((PDIM + BN - 1) / BN), (unsigned)(M / BM)),
                            dim3(256), 0, stream>>>(
        conv_bf, xproj_wb, M, PDIM, IDIM, nullptr,
        xpj, nullptr, EPI_F32, PDIM, 0);

    // 5) ts -> bf16
    ts_extract_kernel<<<dim3((unsigned)(M * RRANK / 256)), dim3(256), 0, stream>>>(xpj, ts_bf);

    // 6) dt GEMM: [M,R]@[R,I] + dt_b -> softplus -> delta (f32)
    gemm_bf16_kernel<false><<<dim3((unsigned)(IDIM / BN), (unsigned)(M / BM)),
                             dim3(256), 0, stream>>>(
        ts_bf, dt_wb, M, IDIM, RRANK, dt_b,
        delta_f, nullptr, EPI_SOFTPLUS, IDIM, 0);

    // 7) sequential scan -> ssm (bf16)
    scan_kernel<<<dim3((unsigned)(BSZ * IDIM * NSTATE / 256)), dim3(256), 0, stream>>>(
        conv_f, delta_f, gate_f, xpj, A_log, Dvec, ssm_bf);

    // 8) out-proj GEMM: [M,I]@[I,H] + out_b -> d_out (f32)
    gemm_bf16_kernel<false><<<dim3((unsigned)(HDIM / BN), (unsigned)(M / BM)),
                             dim3(256), 0, stream>>>(
        ssm_bf, out_wb, M, HDIM, IDIM, out_b,
        out, nullptr, EPI_F32, HDIM, 0);
}
